// Transformer_DTMFormerV2FirstStageChoose_57320633533090
// MI455X (gfx1250) — matrix operand order repaired and checked by reading, not compiled, not run on hardware
//
#include <hip/hip_runtime.h>
#include <math.h>

// ---------------------------------------------------------------------------
// DTMFormerV2 forward for MI455X (gfx1250).
// - All dense projections: bf16 WMMA GEMM (v_wmma_f32_16x16x32_bf16),
//   64x64 block tile, 4 wave32 waves, double-buffered LDS pipeline.
// - Weight tiles streamed by the Tensor Data Mover (tensor_load_to_lds,
//   s_wait_tensorcnt) for the row-major-W path; vector b128 staging otherwise.
// - Irregular steps (softmax attention, DPC-kNN, segment merges, token
//   down/up) are lightweight VALU/atomic kernels (O(B*N^2), bandwidth-trivial).
// All launched GEMMs have M,N % 64 == 0 and K % 32 == 0 (no guards needed).
// ---------------------------------------------------------------------------

typedef __attribute__((ext_vector_type(16))) __bf16 v16bf;
typedef __attribute__((ext_vector_type(16))) unsigned short v16us;
typedef __attribute__((ext_vector_type(8))) float v8f;
typedef __attribute__((ext_vector_type(4))) unsigned int v4u;
typedef __attribute__((ext_vector_type(8))) int v8i;
typedef __attribute__((ext_vector_type(4))) int v4i;

__device__ __forceinline__ unsigned short f2bf(float x){
  unsigned int u = __float_as_uint(x);
  u += 0x7FFFu + ((u >> 16) & 1u);      // round-to-nearest-even
  return (unsigned short)(u >> 16);
}
__device__ __forceinline__ float geluf(float x){
  return 0.5f * x * (1.0f + tanhf(0.7978845608028654f * (x + 0.044715f * x * x * x)));
}
__device__ __forceinline__ void atomicMaxF(float* addr, float val){
  unsigned int* ua = (unsigned int*)addr;
  unsigned int old = *ua;
  while (__uint_as_float(old) < val){
    unsigned int prev = atomicCAS(ua, old, __float_as_uint(val));
    if (prev == old) break;
    old = prev;
  }
}

// Issue a TDM 2D tile load: 32 rows x 64 fp32, row stride ldw (elements),
// into LDS at byte offset lds_off. One call per wave => guard to wave 0.
__device__ __forceinline__ void tdm_load_tile(const float* gsrc, unsigned lds_off, int ldw){
  unsigned long long ga = (unsigned long long)(const void*)gsrc;
  v4u g0;
  g0.x = 1u;                                            // count=1 (valid user D#)
  g0.y = lds_off;                                       // lds_addr
  g0.z = (unsigned)(ga & 0xFFFFFFFFull);                // global_addr[31:0]
  g0.w = (unsigned)((ga >> 32) & 0x1FFFFFFull) | (2u << 30); // addr[56:32] | type=2
  v8i g1;
  g1[0] = (int)(2u << 16);       // workgroup_mask=0, data_size=2 (4 bytes)
  g1[1] = (int)(64u << 16);      // tensor_dim0[15:0]=64 in bits 63:48
  g1[2] = (int)(32u << 16);      // tensor_dim0 hi=0 | tensor_dim1[15:0]=32
  g1[3] = (int)(64u << 16);      // tensor_dim1 hi=0 | tile_dim0=64
  g1[4] = (int)(32u);            // tile_dim1=32, tile_dim2=0
  g1[5] = (int)(unsigned)ldw;    // tensor_dim0_stride[31:0]
  g1[6] = 0;                     // stride0 hi, stride1 lo
  g1[7] = 0;
  v4i z; z[0]=0; z[1]=0; z[2]=0; z[3]=0;
  v8i z8; z8[0]=0; z8[1]=0; z8[2]=0; z8[3]=0; z8[4]=0; z8[5]=0; z8[6]=0; z8[7]=0;
  __builtin_amdgcn_tensor_load_to_lds(g0, g1, z, z, z8, 0);
}

// ---------------- WMMA GEMM ------------------------------------------------
// C[M,N] = act(A[M,K] @ W + bias) + res
// A fp32 row-major (lda=K). W fp32: [K,N] row-major (wtrans=0, TDM path) or
// [N,K] (wtrans=1, vector path), leading dim ldw. Batched via blockIdx.z.
__global__ __launch_bounds__(128) void k_gemm(
    const float* __restrict__ A, const float* __restrict__ W,
    const float* __restrict__ bias, const float* __restrict__ res,
    float* __restrict__ C, int M, int N, int K, int ldw, int wtrans, int doGelu,
    long long sA, long long sW, long long sC)
{
  (void)M;
  A += (long long)blockIdx.z * sA;
  W += (long long)blockIdx.z * sW;
  C += (long long)blockIdx.z * sC;
  const float* resb = res ? res + (long long)blockIdx.z * sC : (const float*)0;

  __shared__ __align__(16) unsigned short As[2][64 * 32]; // bf16 [m][k]
  __shared__ __align__(16) unsigned short Bs[2][64 * 32]; // bf16 [n][k]  (wtrans)
  __shared__ __align__(16) float          Bf[2][32 * 64]; // fp32 [k][n]  (TDM)

  int tid  = threadIdx.x;
  int wave = tid >> 5, lane = tid & 31;
  int half = lane >> 4, lr = lane & 15;
  int row0 = blockIdx.y * 64, col0 = blockIdx.x * 64;
  bool tdm = (wtrans == 0);

  float aR[16];
  float bR[16];

  auto loadA = [&](int k0){
    #pragma unroll
    for (int i = 0; i < 4; i++){
      int q = tid + i * 128;                 // 0..511 float4 slots
      int m = q >> 3, kq = (q & 7) * 4;
      const float4 v = *reinterpret_cast<const float4*>(A + (long long)(row0 + m) * K + k0 + kq);
      aR[4*i+0] = v.x; aR[4*i+1] = v.y; aR[4*i+2] = v.z; aR[4*i+3] = v.w;
    }
  };
  auto storeA = [&](int buf){
    unsigned int* dst = (unsigned int*)&As[buf][0];
    #pragma unroll
    for (int i = 0; i < 4; i++){
      int q = tid + i * 128;
      int m = q >> 3, kq = (q & 7) * 4;
      unsigned p0 = (unsigned)f2bf(aR[4*i+0]) | ((unsigned)f2bf(aR[4*i+1]) << 16);
      unsigned p1 = (unsigned)f2bf(aR[4*i+2]) | ((unsigned)f2bf(aR[4*i+3]) << 16);
      int base = (m * 32 + kq) >> 1;
      dst[base]     = p0;
      dst[base + 1] = p1;
    }
  };
  auto loadB = [&](int k0){                  // wtrans==1: W[n][k]
    #pragma unroll
    for (int i = 0; i < 4; i++){
      int q = tid + i * 128;
      int n = q >> 3, kq = (q & 7) * 4;
      const float4 v = *reinterpret_cast<const float4*>(W + (long long)(col0 + n) * ldw + k0 + kq);
      bR[4*i+0] = v.x; bR[4*i+1] = v.y; bR[4*i+2] = v.z; bR[4*i+3] = v.w;
    }
  };
  auto storeB = [&](int buf){
    unsigned int* dst = (unsigned int*)&Bs[buf][0];
    #pragma unroll
    for (int i = 0; i < 4; i++){
      int q = tid + i * 128;
      int n = q >> 3, kq = (q & 7) * 4;
      unsigned p0 = (unsigned)f2bf(bR[4*i+0]) | ((unsigned)f2bf(bR[4*i+1]) << 16);
      unsigned p1 = (unsigned)f2bf(bR[4*i+2]) | ((unsigned)f2bf(bR[4*i+3]) << 16);
      int base = (n * 32 + kq) >> 1;
      dst[base]     = p0;
      dst[base + 1] = p1;
    }
  };
  auto tdmB = [&](int k0, int buf){          // wtrans==0: stream W tile via TDM
    if (tid < 32)
      tdm_load_tile(W + (long long)k0 * ldw + col0,
                    (unsigned)(unsigned long long)(void*)&Bf[buf][0], ldw);
  };
  auto tdmWait = [&](){ if (tid < 32) __builtin_amdgcn_s_wait_tensorcnt(0); };

  v8f acc[4];
  #pragma unroll
  for (int t = 0; t < 4; t++)
    #pragma unroll
    for (int r = 0; r < 8; r++) acc[t][r] = 0.0f;

  auto compute = [&](int buf){
    const unsigned int* Ap = (const unsigned int*)&As[buf][0];
    v16us afr;
    int mL = wave * 16 + lr;
    #pragma unroll
    for (int p = 0; p < 8; p++){
      int kk = half * 16 + 2 * p;
      unsigned u = Ap[(mL * 32 + kk) >> 1];
      afr[2*p]   = (unsigned short)(u & 0xFFFFu);
      afr[2*p+1] = (unsigned short)(u >> 16);
    }
    if (tdm){
      const float* Bp = &Bf[buf][0];
      #pragma unroll
      for (int t = 0; t < 4; t++){
        v16us bfr;
        int nL = t * 16 + lr;
        #pragma unroll
        for (int p = 0; p < 8; p++){
          int kk = half * 16 + 2 * p;
          bfr[2*p]   = f2bf(Bp[kk * 64 + nL]);
          bfr[2*p+1] = f2bf(Bp[(kk + 1) * 64 + nL]);
        }
        acc[t] = __builtin_amdgcn_wmma_f32_16x16x32_bf16(
                   false, __builtin_bit_cast(v16bf, afr),
                   false, __builtin_bit_cast(v16bf, bfr),
                   (short)0, acc[t], false, false);
      }
    } else {
      const unsigned int* Bp = (const unsigned int*)&Bs[buf][0];
      #pragma unroll
      for (int t = 0; t < 4; t++){
        v16us bfr;
        int nL = t * 16 + lr;
        #pragma unroll
        for (int p = 0; p < 8; p++){
          int kk = half * 16 + 2 * p;
          unsigned u = Bp[(nL * 32 + kk) >> 1];
          bfr[2*p]   = (unsigned short)(u & 0xFFFFu);
          bfr[2*p+1] = (unsigned short)(u >> 16);
        }
        acc[t] = __builtin_amdgcn_wmma_f32_16x16x32_bf16(
                   false, __builtin_bit_cast(v16bf, afr),
                   false, __builtin_bit_cast(v16bf, bfr),
                   (short)0, acc[t], false, false);
      }
    }
  };

  // ---- software pipeline: tile k+1 in flight during WMMAs of tile k ----
  if (tdm) tdmB(0, 0); else loadB(0);
  loadA(0);
  storeA(0);
  if (!tdm) storeB(0); else tdmWait();
  __syncthreads();

  int buf = 0;
  for (int k0 = 0; k0 < K; k0 += 32){
    bool more = (k0 + 32) < K;
    if (more){
      if (tdm) tdmB(k0 + 32, buf ^ 1); else loadB(k0 + 32);
      loadA(k0 + 32);
    }
    compute(buf);
    if (more){
      storeA(buf ^ 1);
      if (!tdm) storeB(buf ^ 1); else tdmWait();
    }
    __syncthreads();
    buf ^= 1;
  }

  // epilogue: D layout: lane=N, VGPR r -> M = r + 8*half
  #pragma unroll
  for (int t = 0; t < 4; t++){
    #pragma unroll
    for (int r = 0; r < 8; r++){
      int m = row0 + wave * 16 + r + half * 8;
      int n = col0 + t * 16 + lr;
      float v = acc[t][r];
      if (bias)  v += bias[n];
      if (doGelu) v = geluf(v);
      if (resb)  v += resb[(long long)m * N + n];
      C[(long long)m * N + n] = v;
    }
  }
}

// ---------------- elementwise / small kernels ------------------------------
__global__ void k_im2col(const float* __restrict__ img, float* __restrict__ A, int total){
  int gid = blockIdx.x * blockDim.x + threadIdx.x;
  if (gid >= total) return;
  int k = gid % 768; int m = gid / 768;
  int b = m >> 8; int t = m & 255; int py = t >> 4, px = t & 15;
  int c = k >> 8; int r = k & 255; int i = r >> 4, j = r & 15;
  A[gid] = img[(((long long)(b * 3 + c) * 256 + py * 16 + i) * 256) + px * 16 + j];
}

__global__ void k_addpos(float* __restrict__ x, const float* __restrict__ pos, int total){
  int gid = blockIdx.x * blockDim.x + threadIdx.x;
  if (gid >= total) return;
  x[gid] += pos[gid % (256 * 768)];
}

__global__ void k_ln(const float* __restrict__ x, const float* __restrict__ g,
                     const float* __restrict__ b, float* __restrict__ y, int M, int C){
  int m = blockIdx.x * blockDim.x + threadIdx.x;
  if (m >= M) return;
  const float* xr = x + (long long)m * C;
  float mu = 0.f;
  for (int c = 0; c < C; c++) mu += xr[c];
  mu /= (float)C;
  float v = 0.f;
  for (int c = 0; c < C; c++){ float d = xr[c] - mu; v += d * d; }
  v /= (float)C;
  float rs = rsqrtf(v + 1e-5f);
  float* yr = y + (long long)m * C;
  for (int c = 0; c < C; c++) yr[c] = (xr[c] - mu) * rs * g[c] + b[c];
}

// one workgroup per (b,h); queries processed sequentially with LDS softmax
__global__ __launch_bounds__(128) void k_attn(const float* __restrict__ qkv,
    float* __restrict__ O, float* __restrict__ colsum, int N, int H, int hd, int wantScore){
  int bh = blockIdx.x; int b = bh / H; int h = bh % H;
  int D = H * hd;
  const float* base = qkv + (long long)b * N * 3 * D;
  __shared__ float sQ[128];
  __shared__ float sS[256];
  __shared__ float sCol[256];
  __shared__ float red[128];
  int tid = threadIdx.x;
  float scale = rsqrtf((float)hd);
  for (int j = tid; j < N; j += 128) sCol[j] = 0.f;
  __syncthreads();
  for (int q = 0; q < N; q++){
    const float* qp = base + (long long)q * 3 * D + h * hd;
    for (int e = tid; e < hd; e += 128) sQ[e] = qp[e];
    __syncthreads();
    for (int j = tid; j < N; j += 128){
      const float* kp = base + (long long)j * 3 * D + D + h * hd;
      float s = 0.f;
      for (int e = 0; e < hd; e++) s += sQ[e] * kp[e];
      sS[j] = s * scale;
    }
    __syncthreads();
    float mx = -3.4e38f;
    for (int j = tid; j < N; j += 128) mx = fmaxf(mx, sS[j]);
    red[tid] = mx; __syncthreads();
    for (int s2 = 64; s2 > 0; s2 >>= 1){ if (tid < s2) red[tid] = fmaxf(red[tid], red[tid + s2]); __syncthreads(); }
    mx = red[0]; __syncthreads();
    float sm = 0.f;
    for (int j = tid; j < N; j += 128){ float e = __expf(sS[j] - mx); sS[j] = e; sm += e; }
    red[tid] = sm; __syncthreads();
    for (int s2 = 64; s2 > 0; s2 >>= 1){ if (tid < s2) red[tid] += red[tid + s2]; __syncthreads(); }
    sm = red[0]; __syncthreads();
    float inv = 1.f / sm;
    for (int j = tid; j < N; j += 128){
      sS[j] *= inv;
      if (wantScore && h < 2) sCol[j] += sS[j];
    }
    __syncthreads();
    float* op = O + ((long long)(b * N + q)) * D + h * hd;
    for (int e = tid; e < hd; e += 128){
      float acc = 0.f;
      const float* vbase = base + 2 * D + h * hd + e;
      for (int j = 0; j < N; j++) acc += sS[j] * vbase[(long long)j * 3 * D];
      op[e] = acc;
    }
    __syncthreads();
  }
  if (wantScore && h < 2){
    for (int j = tid; j < N; j += 128) atomicAdd(&colsum[(long long)b * N + j], sCol[j]);
  }
}

__global__ void k_minmax(const float* __restrict__ m, float* __restrict__ out, int Bn, int N){
  int b = blockIdx.x * blockDim.x + threadIdx.x;
  if (b >= Bn) return;
  const float* r = m + (long long)b * N;
  float mn = 3.4e38f, mx = -3.4e38f;
  for (int i = 0; i < N; i++){ mn = fminf(mn, r[i]); mx = fmaxf(mx, r[i]); }
  float inv = 1.f / (mx - mn + 1e-8f);
  float* o = out + (long long)b * N;
  for (int i = 0; i < N; i++) o[i] = (r[i] - mn) * inv;
}

__global__ void k_fill(float* p, long long n, float v){
  long long gid = (long long)blockIdx.x * blockDim.x + threadIdx.x;
  if (gid < n) p[gid] = v;
}
__global__ void k_copy(const float* s, float* d, long long n){
  long long gid = (long long)blockIdx.x * blockDim.x + threadIdx.x;
  if (gid < n) d[gid] = s[gid];
}
__global__ void k_exp(const float* s, float* w, long long n){
  long long gid = (long long)blockIdx.x * blockDim.x + threadIdx.x;
  if (gid < n) w[gid] = __expf(s[gid]);
}

__global__ void k_sumsq(const float* __restrict__ x, float* __restrict__ x2, int M, int C){
  int m = blockIdx.x * blockDim.x + threadIdx.x;
  if (m >= M) return;
  const float* xr = x + (long long)m * C;
  float s = 0.f;
  for (int c = 0; c < C; c++) s += xr[c] * xr[c];
  x2[m] = s;
}
__global__ void k_distfin(float* __restrict__ dist, const float* __restrict__ x2, int Bn, int N){
  long long gid = (long long)blockIdx.x * blockDim.x + threadIdx.x;
  long long total = (long long)Bn * N * N;
  if (gid >= total) return;
  int j = (int)(gid % N);
  int i = (int)((gid / N) % N);
  int b = (int)(gid / ((long long)N * N));
  dist[gid] = x2[b * N + i] + x2[b * N + j] - 2.0f * dist[gid];
}
__global__ void k_knn(const float* __restrict__ dist, float* __restrict__ density, int Bn, int N){
  int gid = blockIdx.x * blockDim.x + threadIdx.x;
  if (gid >= Bn * N) return;
  int b = gid / N, i = gid % N;
  const float* row = dist + ((long long)b * N + i) * N;
  float nb[5] = {3.4e38f, 3.4e38f, 3.4e38f, 3.4e38f, 3.4e38f};
  for (int j = 0; j < N; j++){
    float d = row[j];
    if (d < nb[4]){
      nb[4] = d;
      for (int t = 4; t > 0; t--){ if (nb[t] < nb[t-1]){ float tmp = nb[t]; nb[t] = nb[t-1]; nb[t-1] = tmp; } }
    }
  }
  float mean = (nb[0] + nb[1] + nb[2] + nb[3] + nb[4]) * 0.2f;
  density[gid] = __expf(-mean) + (float)i * 1e-6f;
}
__global__ void k_rowmax(const float* __restrict__ dist, float* __restrict__ dmax, int Bn, int N){
  int gid = blockIdx.x * blockDim.x + threadIdx.x;
  if (gid >= Bn * N) return;
  int b = gid / N, i = gid % N;
  const float* row = dist + ((long long)b * N + i) * N;
  float mx = -3.4e38f;
  for (int j = 0; j < N; j++) mx = fmaxf(mx, row[j]);
  atomicMaxF(&dmax[b], mx);
}
__global__ void k_parent(const float* __restrict__ density, const float* __restrict__ dist,
                         const float* __restrict__ dmax, float* __restrict__ scoreD, int Bn, int N){
  int gid = blockIdx.x * blockDim.x + threadIdx.x;
  if (gid >= Bn * N) return;
  int b = gid / N, i = gid % N;
  float di = density[b * N + i];
  const float* row = dist + ((long long)b * N + i) * N;
  float best = dmax[b];
  for (int j = 0; j < N; j++){
    if (density[b * N + j] > di) best = fminf(best, row[j]);
  }
  scoreD[gid] = best * di;
}
__global__ void k_topk(const float* __restrict__ scoreD, float* __restrict__ scratch,
                       int* __restrict__ centers, int Bn, int N, int M){
  int b = blockIdx.x * blockDim.x + threadIdx.x;
  if (b >= Bn) return;
  float* s = scratch + (long long)b * N;
  const float* src = scoreD + (long long)b * N;
  for (int i = 0; i < N; i++) s[i] = src[i];
  for (int m = 0; m < M; m++){
    int bi = 0; float bv = s[0];
    for (int i = 1; i < N; i++){ if (s[i] > bv){ bv = s[i]; bi = i; } }
    centers[b * M + m] = bi;
    s[bi] = -3.4e38f;
  }
}
__global__ void k_assign(const float* __restrict__ dist, const int* __restrict__ centers,
                         int* __restrict__ idxc, int Bn, int N, int M){
  int gid = blockIdx.x * blockDim.x + threadIdx.x;
  if (gid >= Bn * N) return;
  int b = gid / N, n = gid % N;
  int bi = 0; float bv = 3.4e38f;
  for (int m = 0; m < M; m++){
    float d = dist[((long long)b * N + centers[b * M + m]) * N + n];
    if (d < bv){ bv = d; bi = m; }
  }
  idxc[gid] = bi;
}
__global__ void k_fixc(const int* __restrict__ centers, int* __restrict__ idxc, int Bn, int N, int M){
  int gid = blockIdx.x * blockDim.x + threadIdx.x;
  if (gid >= Bn * M) return;
  int b = gid / M, m = gid % M;
  idxc[b * N + centers[b * M + m]] = m;
}
__global__ void k_segden(const int* __restrict__ idxc, const float* __restrict__ w,
                         float* __restrict__ denom, int Bn, int N, int M){
  int gid = blockIdx.x * blockDim.x + threadIdx.x;
  if (gid >= Bn * N) return;
  int b = gid / N;
  atomicAdd(&denom[b * M + idxc[gid]], w[gid]);
}
__global__ void k_segacc(const float* __restrict__ X, const int* __restrict__ idxc,
                         const float* __restrict__ w, const float* __restrict__ denom,
                         float* __restrict__ meanA, float* __restrict__ sqA,
                         float* __restrict__ normw, int Bn, int N, int M, int C){
  int gid = blockIdx.x * blockDim.x + threadIdx.x;
  if (gid >= Bn * N) return;
  int b = gid / N;
  int cl = idxc[gid];
  float nw = w[gid] / (denom[b * M + cl] + 1e-6f);
  normw[gid] = nw;
  const float* xr = X + (long long)gid * C;
  float* mr = meanA + ((long long)b * M + cl) * C;
  float* sr = sqA   + ((long long)b * M + cl) * C;
  for (int c = 0; c < C; c++){
    float xv = xr[c];
    atomicAdd(&mr[c], xv * nw);
    atomicAdd(&sr[c], xv * xv * nw);
  }
}
__global__ void k_var(const float* __restrict__ meanA, const float* __restrict__ sqA,
                      float* __restrict__ v, int rows, int C){
  int m = blockIdx.x * blockDim.x + threadIdx.x;
  if (m >= rows) return;
  const float* mr = meanA + (long long)m * C;
  const float* sr = sqA   + (long long)m * C;
  float acc = 0.f;
  for (int c = 0; c < C; c++) acc += sr[c] - mr[c] * mr[c];
  v[m] = acc / (float)C;
}
__global__ void k_htmpost(const int* __restrict__ idxc, const float* __restrict__ normw,
                          const int* __restrict__ prevIdx, const float* __restrict__ prevAgg,
                          int* __restrict__ outIdx, float* __restrict__ outAgg,
                          int Bn, int N0, int Nprev){
  int gid = blockIdx.x * blockDim.x + threadIdx.x;
  if (gid >= Bn * N0) return;
  int b = gid / N0, n0 = gid % N0;
  int t = prevIdx ? prevIdx[gid] : n0;
  outIdx[gid] = idxc[b * Nprev + t];
  float pa = prevAgg ? prevAgg[gid] : 1.0f;
  outAgg[gid] = pa * normw[b * Nprev + t];
}
__global__ void k_scatter(const float* __restrict__ xs, const int* __restrict__ idx_s,
                          const int* __restrict__ idx_t, const float* __restrict__ aggw,
                          float* __restrict__ num, float* __restrict__ den,
                          int Bn, int N0, int Ns, int Nt, int C){
  int gid = blockIdx.x * blockDim.x + threadIdx.x;
  if (gid >= Bn * N0) return;
  int b = gid / N0, n0 = gid % N0;
  int s = idx_s ? idx_s[gid] : n0;
  int t = idx_t ? idx_t[gid] : n0;
  float wgt = aggw ? aggw[gid] : 1.0f;
  const float* f = xs + ((long long)(b * Ns + s)) * C;
  float* nr = num + ((long long)(b * Nt + t)) * C;
  for (int c = 0; c < C; c++) atomicAdd(&nr[c], f[c] * wgt);
  atomicAdd(&den[b * Nt + t], wgt);
}
__global__ void k_div(float* __restrict__ num, const float* __restrict__ den, int rows, int C){
  long long gid = (long long)blockIdx.x * blockDim.x + threadIdx.x;
  long long total = (long long)rows * C;
  if (gid >= total) return;
  int r = (int)(gid / C);
  num[gid] /= (den[r] + 1e-6f);
}
__global__ void k_concat(const float* __restrict__ A1, int C1, const float* __restrict__ A2, int C2,
                         float* __restrict__ out, int Mrows){
  long long gid = (long long)blockIdx.x * blockDim.x + threadIdx.x;
  int Cc = C1 + C2;
  long long total = (long long)Mrows * Cc;
  if (gid >= total) return;
  int c = (int)(gid % Cc);
  long long m = gid / Cc;
  out[gid] = (c < C1) ? A1[m * C1 + c] : A2[m * C2 + (c - C1)];
}
__global__ void k_gatherscore(const float* __restrict__ src, const int* __restrict__ idx,
                              float* __restrict__ out, int Bn, int N0, int Nc){
  int gid = blockIdx.x * blockDim.x + threadIdx.x;
  if (gid >= Bn * N0) return;
  int b = gid / N0;
  out[gid] = src[b * Nc + idx[gid]];
}

// ---------------- host orchestration ---------------------------------------
static inline dim3 g1(long long n, int b){ return dim3((unsigned)((n + b - 1) / b)); }

static void gemm(hipStream_t st, const float* A, const float* W, const float* bias,
                 const float* res, float* C, int M, int N, int K, int ldw, int wtrans,
                 int doGelu, long long sA = 0, long long sW = 0, long long sC = 0, int batch = 1){
  dim3 grid((N + 63) / 64, (M + 63) / 64, batch);
  k_gemm<<<grid, dim3(128), 0, st>>>(A, W, bias, res, C, M, N, K, ldw, wtrans, doGelu, sA, sW, sC);
}

struct Blk { const float *ln1g,*ln1b,*qkvw,*projw,*projb,*ln2g,*ln2b,*fc1w,*fc1b,*fc2w,*fc2b; };

static void runBlock(hipStream_t st, float* X, int Bn, int N, int d, int H, const Blk& p,
                     float* Y, float* QKV, float* O, float* Hbuf, float* colsum, int wantScore){
  int M = Bn * N; int hd = d / H;
  k_ln<<<g1(M, 256), 256, 0, st>>>(X, p.ln1g, p.ln1b, Y, M, d);
  gemm(st, Y, p.qkvw, nullptr, nullptr, QKV, M, 3 * d, d, 3 * d, 0, 0);
  if (wantScore) k_fill<<<g1((long long)Bn * N, 256), 256, 0, st>>>(colsum, (long long)Bn * N, 0.f);
  k_attn<<<dim3(Bn * H), dim3(128), 0, st>>>(QKV, O, colsum, N, H, hd, wantScore);
  gemm(st, O, p.projw, p.projb, X, X, M, d, d, d, 0, 0);
  k_ln<<<g1(M, 256), 256, 0, st>>>(X, p.ln2g, p.ln2b, Y, M, d);
  gemm(st, Y, p.fc1w, p.fc1b, nullptr, Hbuf, M, 4 * d, d, 4 * d, 0, 1);
  gemm(st, Hbuf, p.fc2w, p.fc2b, X, X, M, d, 4 * d, d, 0, 0);
}

// DPC-kNN cluster + weighted merge. Writes idxc, normw, meanA (merged x), varOut.
static void runHTM(hipStream_t st, const float* X, int Bn, int N, int C, int M,
                   float* dist, float* x2, float* density, float* scoreD, float* scratch,
                   float* dmax, int* centers, int* idxc, const float* scoreNorm, float* tokenw,
                   float* denom, float* normw, float* meanA, float* sqA, float* varOut){
  k_sumsq<<<g1(Bn * N, 256), 256, 0, st>>>(X, x2, Bn * N, C);
  gemm(st, X, X, nullptr, nullptr, dist, N, N, C, C, 1, 0,
       (long long)N * C, (long long)N * C, (long long)N * N, Bn);
  k_distfin<<<g1((long long)Bn * N * N, 256), 256, 0, st>>>(dist, x2, Bn, N);
  k_knn<<<g1(Bn * N, 256), 256, 0, st>>>(dist, density, Bn, N);
  k_fill<<<g1(Bn, 256), 256, 0, st>>>(dmax, Bn, -3.4e38f);
  k_rowmax<<<g1(Bn * N, 256), 256, 0, st>>>(dist, dmax, Bn, N);
  k_parent<<<g1(Bn * N, 256), 256, 0, st>>>(density, dist, dmax, scoreD, Bn, N);
  k_topk<<<dim3(1), dim3(64), 0, st>>>(scoreD, scratch, centers, Bn, N, M);
  k_assign<<<g1(Bn * N, 256), 256, 0, st>>>(dist, centers, idxc, Bn, N, M);
  k_fixc<<<g1(Bn * M, 256), 256, 0, st>>>(centers, idxc, Bn, N, M);
  k_exp<<<g1((long long)Bn * N, 256), 256, 0, st>>>(scoreNorm, tokenw, (long long)Bn * N);
  k_fill<<<g1((long long)Bn * M, 256), 256, 0, st>>>(denom, (long long)Bn * M, 0.f);
  k_segden<<<g1(Bn * N, 256), 256, 0, st>>>(idxc, tokenw, denom, Bn, N, M);
  k_fill<<<g1((long long)Bn * M * C, 256), 256, 0, st>>>(meanA, (long long)Bn * M * C, 0.f);
  k_fill<<<g1((long long)Bn * M * C, 256), 256, 0, st>>>(sqA, (long long)Bn * M * C, 0.f);
  k_segacc<<<g1(Bn * N, 256), 256, 0, st>>>(X, idxc, tokenw, denom, meanA, sqA, normw, Bn, N, M, C);
  k_var<<<g1(Bn * M, 256), 256, 0, st>>>(meanA, sqA, varOut, Bn * M, C);
}

extern "C" void kernel_launch(void* const* d_in, const int* in_sizes, int n_in,
                              void* d_out, int out_size, void* d_ws, size_t ws_size,
                              hipStream_t stream){
  (void)in_sizes; (void)n_in; (void)out_size; (void)ws_size;
  const int Bn = 32, N0 = 256, N1 = 64, N2 = 16;
  const int D0 = 192, D1 = 384, D2 = 768;
  const int H0 = 4, H1 = 6, H2 = 8;

  auto F = [&](int i){ return (const float*)d_in[i]; };
  int ip = 1;
  const float* patch_w = F(ip++);  const float* patch_b = F(ip++);
  const float* pos     = F(ip++);
  const float* to_pe_w = F(ip++);  const float* to_pe_b = F(ip++);
  const float* fr_pe_w = F(ip++);  const float* fr_pe_b = F(ip++);
  auto nextBlk = [&](){
    Blk b;
    b.ln1g = F(ip++); b.ln1b = F(ip++); b.qkvw = F(ip++);
    b.projw = F(ip++); b.projb = F(ip++);
    b.ln2g = F(ip++); b.ln2b = F(ip++);
    b.fc1w = F(ip++); b.fc1b = F(ip++); b.fc2w = F(ip++); b.fc2b = F(ip++);
    return b;
  };
  Blk s0[2]; for (int i = 0; i < 2; i++) s0[i] = nextBlk();
  const float* n0g = F(ip++); const float* n0b = F(ip++);
  const float* atm1w = F(ip++); const float* atm1b = F(ip++);
  Blk s1[2]; for (int i = 0; i < 2; i++) s1[i] = nextBlk();
  const float* n1g = F(ip++); const float* n1b = F(ip++);
  const float* atm2w = F(ip++); const float* atm2b = F(ip++);
  Blk s2[4]; for (int i = 0; i < 4; i++) s2[i] = nextBlk();
  const float* n2g = F(ip++); const float* n2b = F(ip++);
  const float* ltr3w = F(ip++); const float* ltr3b = F(ip++);
  Blk s3[2]; for (int i = 0; i < 2; i++) s3[i] = nextBlk();
  const float* n3g = F(ip++); const float* n3b = F(ip++);
  const float* ltr4w = F(ip++); const float* ltr4b = F(ip++);
  Blk s4[2]; for (int i = 0; i < 2; i++) s4[i] = nextBlk();
  const float* n4g = F(ip++); const float* n4b = F(ip++);

  // output regions (flat, reference return order)
  float* out     = (float*)d_out;
  float* out_enc = out;                              // 32*256*768
  float* out_as0 = out_enc + (long long)Bn * N0 * 768;
  float* out_as1 = out_as0 + Bn * N0;
  float* out_s1  = out_as1 + Bn * N0;
  float* out_s2  = out_s1 + Bn * N0;
  float* out_v1  = out_s2 + Bn * N1;
  float* out_v2  = out_v1 + Bn * N1;

  // workspace bump allocator (floats)
  char* wsb = (char*)d_ws; size_t off = 0;
  auto alloc = [&](size_t nf){ void* p = wsb + off; off = (off + nf * 4 + 255) & ~(size_t)255; return (float*)p; };
  float* ABIG  = alloc(8192ull * 768);   // im2col / MLP hidden / concat
  float* QKVB  = alloc(8192ull * 768);   // x768, then qkv per block
  float* X0    = alloc(8192ull * 192);
  float* Ybuf  = alloc(8192ull * 192);
  float* Obuf  = alloc(8192ull * 192);
  float* X1    = alloc(2048ull * 384);
  float* X2    = alloc(512ull * 768);
  float* X3    = alloc(2048ull * 384);
  float* X4    = alloc(8192ull * 192);
  float* DIST  = alloc(32ull * 256 * 256);
  float* NUM   = alloc(8192ull * 384);   // covers 2048*768 too
  float* DEN   = alloc(8192);
  float* x2b   = alloc(8192);
  float* density = alloc(8192);
  float* scoreD  = alloc(8192);
  float* scrS    = alloc(8192);
  float* dmax    = alloc(32);
  float* colsum  = alloc(8192);
  float* score0  = alloc(8192);
  float* score1s = alloc(2048);
  float* tokenw  = alloc(8192);
  float* denom   = alloc(2048);
  float* normw   = alloc(8192);
  float* meanA   = alloc(32ull * 64 * 192);
  float* sqA     = alloc(32ull * 64 * 192);
  float* agg1    = alloc(8192);
  float* agg2    = alloc(8192);
  int* idxc    = (int*)alloc(8192);
  int* centers = (int*)alloc(2048);
  int* idx1    = (int*)alloc(8192);
  int* idx2    = (int*)alloc(8192);

  hipStream_t st = stream;
  // --- patch embed: im2col + GEMM (W transposed) + pos + to_pe ---
  k_im2col<<<g1(8192ll * 768, 256), 256, 0, st>>>((const float*)d_in[0], ABIG, 8192 * 768);
  gemm(st, ABIG, patch_w, patch_b, nullptr, QKVB, 8192, 768, 768, 768, 1, 0);
  k_addpos<<<g1(8192ll * 768, 256), 256, 0, st>>>(QKVB, pos, 8192 * 768);
  gemm(st, QKVB, to_pe_w, to_pe_b, nullptr, X0, 8192, D0, 768, D0, 0, 0);

  // --- stage 0 ---
  for (int i = 0; i < 2; i++)
    runBlock(st, X0, Bn, N0, D0, H0, s0[i], Ybuf, QKVB, Obuf, ABIG, colsum, i == 1);
  k_ln<<<g1(8192, 256), 256, 0, st>>>(X0, n0g, n0b, X0, 8192, D0);
  k_minmax<<<g1(Bn, 32), 32, 0, st>>>(colsum, score0, Bn, N0);
  k_copy<<<g1(Bn * N0, 256), 256, 0, st>>>(score0, out_as0, Bn * N0);
  k_copy<<<g1(Bn * N0, 256), 256, 0, st>>>(score0, out_s1, Bn * N0);

  // --- HTM 1: 256 -> 64, 192 -> 384 ---
  runHTM(st, X0, Bn, N0, D0, N1, DIST, x2b, density, scoreD, scrS, dmax, centers, idxc,
         score0, tokenw, denom, normw, meanA, sqA, out_v1);
  k_htmpost<<<g1(Bn * N0, 256), 256, 0, st>>>(idxc, normw, nullptr, nullptr, idx1, agg1, Bn, N0, N0);
  gemm(st, meanA, atm1w, atm1b, nullptr, X1, Bn * N1, D1, D0, D1, 0, 0);

  // --- stage 1 ---
  for (int i = 0; i < 2; i++)
    runBlock(st, X1, Bn, N1, D1, H1, s1[i], Ybuf, QKVB, Obuf, ABIG, colsum, i == 1);
  k_ln<<<g1(2048, 256), 256, 0, st>>>(X1, n1g, n1b, X1, 2048, D1);
  k_minmax<<<g1(Bn, 32), 32, 0, st>>>(colsum, score1s, Bn, N1);
  k_copy<<<g1(Bn * N1, 256), 256, 0, st>>>(score1s, out_s2, Bn * N1);
  k_gatherscore<<<g1(Bn * N0, 256), 256, 0, st>>>(score1s, idx1, out_as1, Bn, N0, N1);

  // --- HTM 2: 64 -> 16, 384 -> 768 ---
  runHTM(st, X1, Bn, N1, D1, N2, DIST, x2b, density, scoreD, scrS, dmax, centers, idxc,
         score1s, tokenw, denom, normw, meanA, sqA, out_v2);
  k_htmpost<<<g1(Bn * N0, 256), 256, 0, st>>>(idxc, normw, idx1, agg1, idx2, agg2, Bn, N0, N1);
  gemm(st, meanA, atm2w, atm2b, nullptr, X2, Bn * N2, D2, D1, D2, 0, 0);

  // --- stage 2 ---
  for (int i = 0; i < 4; i++)
    runBlock(st, X2, Bn, N2, D2, H2, s2[i], Ybuf, QKVB, Obuf, ABIG, colsum, 0);
  k_ln<<<g1(512, 256), 256, 0, st>>>(X2, n2g, n2b, X2, 512, D2);

  // --- ltr3: restore to stage-1 resolution ---
  k_fill<<<g1((long long)Bn * N1 * D2, 256), 256, 0, st>>>(NUM, (long long)Bn * N1 * D2, 0.f);
  k_fill<<<g1(Bn * N1, 256), 256, 0, st>>>(DEN, Bn * N1, 0.f);
  k_scatter<<<g1(Bn * N0, 256), 256, 0, st>>>(X2, idx2, idx1, agg1, NUM, DEN, Bn, N0, N2, N1, D2);
  k_div<<<g1((long long)Bn * N1 * D2, 256), 256, 0, st>>>(NUM, DEN, Bn * N1, D2);
  k_concat<<<g1((long long)Bn * N1 * (D1 + D2), 256), 256, 0, st>>>(X1, D1, NUM, D2, ABIG, Bn * N1);
  gemm(st, ABIG, ltr3w, ltr3b, nullptr, X3, Bn * N1, D1, D1 + D2, D1, 0, 0);

  // --- stage 3 ---
  for (int i = 0; i < 2; i++)
    runBlock(st, X3, Bn, N1, D1, H1, s3[i], Ybuf, QKVB, Obuf, ABIG, colsum, 0);
  k_ln<<<g1(2048, 256), 256, 0, st>>>(X3, n3g, n3b, X3, 2048, D1);

  // --- ltr4: restore to full resolution ---
  k_fill<<<g1((long long)Bn * N0 * D1, 256), 256, 0, st>>>(NUM, (long long)Bn * N0 * D1, 0.f);
  k_fill<<<g1(Bn * N0, 256), 256, 0, st>>>(DEN, Bn * N0, 0.f);
  k_scatter<<<g1(Bn * N0, 256), 256, 0, st>>>(X3, idx1, nullptr, nullptr, NUM, DEN, Bn, N0, N1, N0, D1);
  k_div<<<g1((long long)Bn * N0 * D1, 256), 256, 0, st>>>(NUM, DEN, Bn * N0, D1);
  k_concat<<<g1((long long)Bn * N0 * (D0 + D1), 256), 256, 0, st>>>(X0, D0, NUM, D1, ABIG, Bn * N0);
  gemm(st, ABIG, ltr4w, ltr4b, nullptr, X4, Bn * N0, D0, D0 + D1, D0, 0, 0);

  // --- stage 4 ---
  for (int i = 0; i < 2; i++)
    runBlock(st, X4, Bn, N0, D0, H0, s4[i], Ybuf, QKVB, Obuf, ABIG, colsum, 0);
  k_ln<<<g1(8192, 256), 256, 0, st>>>(X4, n4g, n4b, X4, 8192, D0);

  // --- final projection to encoded output ---
  gemm(st, X4, fr_pe_w, fr_pe_b, nullptr, out_enc, Bn * N0, 768, D0, 768, 0, 0);
}
